// MultiHeadAttention_55989193671154
// MI455X (gfx1250) — matrix operand order repaired and checked by reading, not compiled, not run on hardware
//
#include <hip/hip_runtime.h>
#include <stdint.h>

// ---------------------------------------------------------------------------
// CDNA5 (gfx1250) wave32 WMMA multi-head attention.
//   prep:  X (f32) -> bf16; W (f32, KxN) -> Wt (bf16, NxK, transposed)
//   proj:  Q = (Xq@Wq + bq) * 1/sqrt(D)  (bf16), K/V likewise (bf16)
//   attn:  flash-attention per (n, head) (bf16)
//   out:   O = A@Wo + bo (f32)
// GEMMs double-buffer LDS tiles with global_load_async_to_lds_b128 (ASYNCcnt).
// ---------------------------------------------------------------------------

typedef __attribute__((ext_vector_type(16))) __bf16 v16bf;
typedef __attribute__((ext_vector_type(8)))  float  v8f;

union FragU {
  v16bf v;
  uint4 q[2];
  unsigned short us[16];
};

__device__ __forceinline__ unsigned short f2bf(float x) {
  union { float f; unsigned int u; } c;
  c.f = x;
  unsigned int u = c.u;
  u += 0x7fffu + ((u >> 16) & 1u);   // round-to-nearest-even
  return (unsigned short)(u >> 16);
}

__device__ __forceinline__ v8f wmma_bf16(v16bf a, v16bf b, v8f c) {
  return __builtin_amdgcn_wmma_f32_16x16x32_bf16(false, a, false, b,
                                                 (short)0, c, false, false);
}

// Per-lane async global->LDS 16-byte copy (tracked by ASYNCcnt).
// LDS offset = low 32 bits of the flat pointer (LDS aperture maps addr[31:0]).
__device__ __forceinline__ void async_b128(const void* g, void* l) {
  asm volatile("global_load_async_to_lds_b128 %0, %1, off"
               :: "v"((unsigned)(size_t)l), "v"((unsigned long long)g)
               : "memory");
}

// ---------------------------------------------------------------------------
// Prep: flat f32 -> bf16 (8 elems / thread)
// ---------------------------------------------------------------------------
__global__ __launch_bounds__(256) void cvt_bf16_kernel(
    const float* __restrict__ in, unsigned short* __restrict__ out, int n) {
  int i = (blockIdx.x * 256 + threadIdx.x) * 8;
  if (i + 8 <= n) {
    float4 a = *(const float4*)(in + i);
    float4 b = *(const float4*)(in + i + 4);
    union { uint4 q; unsigned short us[8]; } pk;
    pk.us[0] = f2bf(a.x); pk.us[1] = f2bf(a.y);
    pk.us[2] = f2bf(a.z); pk.us[3] = f2bf(a.w);
    pk.us[4] = f2bf(b.x); pk.us[5] = f2bf(b.y);
    pk.us[6] = f2bf(b.z); pk.us[7] = f2bf(b.w);
    *(uint4*)(out + i) = pk.q;
  }
}

// ---------------------------------------------------------------------------
// Prep: W[K][N] f32 -> Wt[N][K] bf16 (32x32 tiles via LDS)
// ---------------------------------------------------------------------------
__global__ __launch_bounds__(256) void transpose_bf16_kernel(
    const float* __restrict__ W, unsigned short* __restrict__ Wt, int K, int N) {
  __shared__ float tile[32][33];
  const int tx = threadIdx.x & 31;
  const int ty = threadIdx.x >> 5;     // 0..7
  const int k0 = blockIdx.y * 32;
  const int n0 = blockIdx.x * 32;
  #pragma unroll
  for (int j = 0; j < 4; ++j)
    tile[ty + 8*j][tx] = W[(size_t)(k0 + ty + 8*j) * N + n0 + tx];
  __syncthreads();
  #pragma unroll
  for (int j = 0; j < 4; ++j)
    Wt[(size_t)(n0 + ty + 8*j) * K + k0 + tx] = f2bf(tile[tx][ty + 8*j]);
}

// ---------------------------------------------------------------------------
// GEMM: Out[M,N] = A[M,K](bf16) @ Bt[N,K]^T(bf16) (+bias)*scale
// 256 threads = 8 waves; 128x128 tile; K-step 32; double-buffered LDS with
// async-to-LDS staging. Wave grid 4(M) x 2(N): each wave 32x64 = 2x4 WMMAs.
// ---------------------------------------------------------------------------
template <bool OUT_BF16>
__global__ __launch_bounds__(256) void gemm_tt_kernel(
    const unsigned short* __restrict__ A,
    const unsigned short* __restrict__ Bt,
    const float* __restrict__ bias, void* __restrict__ Out,
    int M, int N, int K, float scale)
{
  __shared__ alignas(16) unsigned short sA[2][128][40];  // [m][k], pad 40
  __shared__ alignas(16) unsigned short sB[2][128][40];  // [n][k], pad 40

  const int tid  = threadIdx.x;
  const int wave = tid >> 5;
  const int lane = tid & 31;
  const int half = lane >> 4;
  const int lr   = lane & 15;

  const int bm = blockIdx.y * 128;
  const int bn = blockIdx.x * 128;

  const int wMb = (wave & 3) * 32;
  const int wNb = (wave >> 2) * 64;

  // chunk ids: 512 16-byte chunks per 128x32 tile, 2 per thread
  const int r0 = tid >> 2,        s0 = tid & 3;
  const int r1 = (tid + 256) >> 2, s1 = (tid + 256) & 3;

  v8f acc[2][4] = {};

  auto stage = [&](int ktile, int buf) {
    const int k0 = ktile * 32;
    async_b128(A  + (size_t)(bm + r0) * K + k0 + s0 * 8, &sA[buf][r0][s0 * 8]);
    async_b128(Bt + (size_t)(bn + r0) * K + k0 + s0 * 8, &sB[buf][r0][s0 * 8]);
    async_b128(A  + (size_t)(bm + r1) * K + k0 + s1 * 8, &sA[buf][r1][s1 * 8]);
    async_b128(Bt + (size_t)(bn + r1) * K + k0 + s1 * 8, &sB[buf][r1][s1 * 8]);
  };

  const int nk = K / 32;
  stage(0, 0);
  for (int i = 0; i < nk; ++i) {
    const int buf = i & 1;
    if (i + 1 < nk) {
      stage(i + 1, buf ^ 1);
      // 4 newly issued outstanding; <=4 left means tile i has landed
      asm volatile("s_wait_asynccnt 0x4" ::: "memory");
    } else {
      asm volatile("s_wait_asynccnt 0x0" ::: "memory");
    }
    __syncthreads();

    FragU a[2];
    #pragma unroll
    for (int im = 0; im < 2; ++im) {
      const unsigned short* p = &sA[buf][wMb + im*16 + lr][half * 8];
      a[im].q[0] = *(const uint4*)p;
      a[im].q[1] = *(const uint4*)(p + 16);
    }
    #pragma unroll
    for (int jn = 0; jn < 4; ++jn) {
      FragU b;
      const unsigned short* p = &sB[buf][wNb + jn*16 + lr][half * 8];
      b.q[0] = *(const uint4*)p;
      b.q[1] = *(const uint4*)(p + 16);
      #pragma unroll
      for (int im = 0; im < 2; ++im)
        acc[im][jn] = wmma_bf16(a[im].v, b.v, acc[im][jn]);
    }
    __syncthreads();
  }

  #pragma unroll
  for (int im = 0; im < 2; ++im) {
    #pragma unroll
    for (int jn = 0; jn < 4; ++jn) {
      int col = bn + wNb + jn * 16 + lr;
      float bv = bias[col];
      #pragma unroll
      for (int r = 0; r < 8; ++r) {
        int row = bm + wMb + im * 16 + half * 8 + r;
        float val = (acc[im][jn][r] + bv) * scale;
        if (OUT_BF16)
          ((unsigned short*)Out)[(size_t)row * N + col] = f2bf(val);
        else
          ((float*)Out)[(size_t)row * N + col] = val;
      }
    }
  }
}

// ---------------------------------------------------------------------------
// Flash attention: grid (S/128, N*H), 256 threads = 8 waves.
// Wave w owns 16 query rows; streams K/V in 128-row tiles via LDS.
// Q pre-scaled by 1/sqrt(D). Tensors (N,S,D) bf16, head slice h*64.
// ---------------------------------------------------------------------------
__global__ __launch_bounds__(256) void attn_kernel(
    const unsigned short* __restrict__ Qb,
    const unsigned short* __restrict__ Kb,
    const unsigned short* __restrict__ Vb,
    unsigned short* __restrict__ Ob,
    int S, int D, int H, int HD)
{
  __shared__ alignas(16) unsigned short sK[128][72];      // [k][d]
  __shared__ alignas(16) unsigned short sV[64][136];      // [d][k] transposed
  __shared__ alignas(16) unsigned short sP[8][16][136];   // per-wave P tile

  const int tid  = threadIdx.x;
  const int wave = tid >> 5;
  const int lane = tid & 31;
  const int half = lane >> 4;
  const int lr   = lane & 15;

  const int n = blockIdx.y / H;
  const int h = blockIdx.y % H;
  const size_t base = ((size_t)n * S) * D + (size_t)h * HD;

  const int qr0 = blockIdx.x * 128 + wave * 16;

  FragU aq[2];
  {
    const unsigned short* qrow = Qb + base + (size_t)(qr0 + lr) * D;
    #pragma unroll
    for (int j = 0; j < 2; ++j) {
      aq[j].q[0] = *(const uint4*)(qrow + 32*j + half*8);
      aq[j].q[1] = *(const uint4*)(qrow + 32*j + half*8 + 16);
    }
  }

  v8f acc[4] = {};
  float mrow[8], lrow[8];
  #pragma unroll
  for (int r = 0; r < 8; ++r) { mrow[r] = -1e30f; lrow[r] = 0.f; }

  const int nkt = S / 128;
  for (int kt = 0; kt < nkt; ++kt) {
    if (kt + 1 < nkt) {
      __builtin_prefetch(Kb + base + (size_t)((kt + 1) * 128) * D, 0, 0);
      __builtin_prefetch(Vb + base + (size_t)((kt + 1) * 128) * D, 0, 0);
    }
    // ---- K tile 128x64: async global->LDS copies ----
    #pragma unroll
    for (int i = 0; i < 4; ++i) {
      int c = tid + 256 * i;
      int row = c >> 3, seg = c & 7;
      async_b128(Kb + base + (size_t)(kt*128 + row) * D + seg*8,
                 &sK[row][seg*8]);
    }
    // ---- V tile transposed -> sV[d][k] (VALU transpose) ----
    #pragma unroll
    for (int i = 0; i < 4; ++i) {
      int c = tid + 256 * i;
      int row = c >> 3, seg = c & 7;
      union { uint4 q; unsigned short us[8]; } vv;
      vv.q = *(const uint4*)(Vb + base + (size_t)(kt*128 + row) * D + seg*8);
      #pragma unroll
      for (int e = 0; e < 8; ++e) sV[seg*8 + e][row] = vv.us[e];
    }
    asm volatile("s_wait_asynccnt 0x0" ::: "memory");
    __syncthreads();

    // ---- logits 16x128 ----
    v8f sc[8];
    #pragma unroll
    for (int ct = 0; ct < 8; ++ct) {
      v8f s = {};
      #pragma unroll
      for (int j = 0; j < 2; ++j) {
        FragU bk;
        const unsigned short* p = &sK[ct*16 + lr][32*j + half*8];
        bk.q[0] = *(const uint4*)p;
        bk.q[1] = *(const uint4*)(p + 16);
        s = wmma_bf16(aq[j].v, bk.v, s);
      }
      sc[ct] = s;
    }

    // ---- online softmax (C-layout row = half*8 + r) ----
    #pragma unroll
    for (int r = 0; r < 8; ++r) {
      float t = sc[0][r];
      #pragma unroll
      for (int ct = 1; ct < 8; ++ct) t = fmaxf(t, sc[ct][r]);
      #pragma unroll
      for (int off = 8; off >= 1; off >>= 1)
        t = fmaxf(t, __shfl_xor(t, off));
      float nm = fmaxf(mrow[r], t);
      float al = __expf(mrow[r] - nm);
      mrow[r] = nm;
      lrow[r] *= al;
      #pragma unroll
      for (int dt = 0; dt < 4; ++dt) acc[dt][r] *= al;
    }
    #pragma unroll
    for (int r = 0; r < 8; ++r) {
      float ts = 0.f;
      #pragma unroll
      for (int ct = 0; ct < 8; ++ct) {
        float e = __expf(sc[ct][r] - mrow[r]);
        sc[ct][r] = e;
        ts += e;
      }
      #pragma unroll
      for (int off = 8; off >= 1; off >>= 1)
        ts += __shfl_xor(ts, off);
      lrow[r] += ts;
    }

    // ---- C-layout -> A-layout via private LDS round trip ----
    #pragma unroll
    for (int ct = 0; ct < 8; ++ct) {
      #pragma unroll
      for (int r = 0; r < 8; ++r)
        sP[wave][half*8 + r][ct*16 + lr] = f2bf(sc[ct][r]);
    }

    // ---- acc += P @ V ----
    #pragma unroll
    for (int kk = 0; kk < 4; ++kk) {
      FragU ap;
      const unsigned short* pp = &sP[wave][lr][kk*32 + half*8];
      ap.q[0] = *(const uint4*)pp;
      ap.q[1] = *(const uint4*)(pp + 16);
      #pragma unroll
      for (int dt = 0; dt < 4; ++dt) {
        FragU bv;
        const unsigned short* pv = &sV[dt*16 + lr][kk*32 + half*8];
        bv.q[0] = *(const uint4*)pv;
        bv.q[1] = *(const uint4*)(pv + 16);
        acc[dt] = wmma_bf16(ap.v, bv.v, acc[dt]);
      }
    }
    __syncthreads();
  }

  #pragma unroll
  for (int r = 0; r < 8; ++r) {
    float inv = 1.0f / lrow[r];
    int row = qr0 + half * 8 + r;
    #pragma unroll
    for (int dt = 0; dt < 4; ++dt)
      Ob[base + (size_t)row * D + dt*16 + lr] = f2bf(acc[dt][r] * inv);
  }
}

// ---------------------------------------------------------------------------
extern "C" void kernel_launch(void* const* d_in, const int* in_sizes, int n_in,
                              void* d_out, int out_size, void* d_ws, size_t ws_size,
                              hipStream_t stream) {
  (void)in_sizes; (void)n_in; (void)out_size; (void)ws_size;

  const float* q_in = (const float*)d_in[0];
  const float* k_in = (const float*)d_in[1];
  const float* v_in = (const float*)d_in[2];
  const float* Wq = (const float*)d_in[3];
  const float* bq = (const float*)d_in[4];
  const float* Wk = (const float*)d_in[5];
  const float* bk = (const float*)d_in[6];
  const float* Wv = (const float*)d_in[7];
  const float* bv = (const float*)d_in[8];
  const float* Wo = (const float*)d_in[9];
  const float* bo = (const float*)d_in[10];

  const int NB = 2, S = 2048, D = 1024, H = 16, HD = 64;
  const int M = NB * S;                 // 4096
  const float scale = 0.03125f;         // 1/sqrt(1024)
  const size_t MD = (size_t)M * D;      // 4M elems
  const size_t DD = (size_t)D * D;      // 1M elems

  unsigned short* p = (unsigned short*)d_ws;
  unsigned short* Xq  = p; p += MD;
  unsigned short* Xk  = p; p += MD;
  unsigned short* Xv  = p; p += MD;
  unsigned short* Wqt = p; p += DD;
  unsigned short* Wkt = p; p += DD;
  unsigned short* Wvt = p; p += DD;
  unsigned short* Wot = p; p += DD;
  unsigned short* Qb  = p; p += MD;
  unsigned short* Kb  = p; p += MD;
  unsigned short* Vb  = p; p += MD;
  unsigned short* Ab  = p; p += MD;

  // prep: convert activations, transpose+convert weights (one-time, L2-resident)
  cvt_bf16_kernel<<<(int)(MD / (256 * 8)), 256, 0, stream>>>(q_in, Xq, (int)MD);
  cvt_bf16_kernel<<<(int)(MD / (256 * 8)), 256, 0, stream>>>(k_in, Xk, (int)MD);
  cvt_bf16_kernel<<<(int)(MD / (256 * 8)), 256, 0, stream>>>(v_in, Xv, (int)MD);
  dim3 gt(D / 32, D / 32);
  transpose_bf16_kernel<<<gt, 256, 0, stream>>>(Wq, Wqt, D, D);
  transpose_bf16_kernel<<<gt, 256, 0, stream>>>(Wk, Wkt, D, D);
  transpose_bf16_kernel<<<gt, 256, 0, stream>>>(Wv, Wvt, D, D);
  transpose_bf16_kernel<<<gt, 256, 0, stream>>>(Wo, Wot, D, D);

  dim3 gg(D / 128, M / 128);            // (8, 32)
  gemm_tt_kernel<true><<<gg, 256, 0, stream>>>(Xq, Wqt, bq, Qb, M, D, D, scale);
  gemm_tt_kernel<true><<<gg, 256, 0, stream>>>(Xk, Wkt, bk, Kb, M, D, D, 1.0f);
  gemm_tt_kernel<true><<<gg, 256, 0, stream>>>(Xv, Wvt, bv, Vb, M, D, D, 1.0f);

  attn_kernel<<<dim3(S / 128, NB * H), 256, 0, stream>>>(Qb, Kb, Vb, Ab, S, D, H, HD);

  gemm_tt_kernel<false><<<gg, 256, 0, stream>>>(Ab, Wot, bo, (float*)d_out, M, D, D, 1.0f);
}
